// TransformerBackbone_64948495450456
// MI455X (gfx1250) — matrix operand order repaired and checked by reading, not compiled, not run on hardware
//
#include <hip/hip_runtime.h>
#include <math.h>

typedef unsigned short ushort_t;
typedef __attribute__((ext_vector_type(16))) __bf16 v16bf;
typedef __attribute__((ext_vector_type(8)))  float  v8f;

#define NSEQ   2048
#define DMODEL 512
#define NHEAD  8
#define HDIM   64
#define NLAYER 2
#define MROWS  4096   /* B*N */
#define DFF    2048

// ---------- helpers ----------
__device__ __forceinline__ ushort_t f2bf(float f) {
  unsigned u = __float_as_uint(f);
  unsigned r = (u + 0x7FFFu + ((u >> 16) & 1u)) >> 16;   // RNE
  return (ushort_t)r;
}

__device__ __forceinline__ float gelu_new(float x) {
  const float c = 0.7978845608028654f;  // sqrt(2/pi)
  float u = c * (x + 0.044715f * x * x * x);
  return 0.5f * x * (1.0f + tanhf(u));
}

__device__ __forceinline__ v8f zero8() {
  v8f z;
  #pragma unroll
  for (int i = 0; i < 8; i++) z[i] = 0.0f;
  return z;
}

// pack two contiguous 8-halfword groups into a v16bf operand
__device__ __forceinline__ v16bf ld16(const ushort_t* lo, const ushort_t* hi) {
  union { v16bf v; uint4 q[2]; } u;
  u.q[0] = *(const uint4*)lo;
  u.q[1] = *(const uint4*)hi;
  return u.v;
}

__device__ __forceinline__ v8f wmma_bf16(v16bf a, v16bf b, v8f c) {
  return __builtin_amdgcn_wmma_f32_16x16x32_bf16(
      false, a, false, b, (short)0, c, false, false);
}

// ---------- small kernels ----------
__global__ void k_cvt_bf16(const float* __restrict__ s, ushort_t* __restrict__ d, int n) {
  int i = blockIdx.x * 256 + threadIdx.x;
  if (i < n) d[i] = f2bf(s[i]);
}

__global__ void k_addpos(const float* __restrict__ e, const float* __restrict__ p,
                         float* __restrict__ h) {
  int i = blockIdx.x * 256 + threadIdx.x;            // MROWS*DMODEL total
  int nd = i & (NSEQ * DMODEL - 1);                  // pos repeats per batch
  h[i] = e[i] + p[nd];
}

__global__ void k_copy(const float* __restrict__ s, float* __restrict__ d, int n) {
  int i = blockIdx.x * 256 + threadIdx.x;
  if (i < n) d[i] = s[i];
}

// one block (256 thr) per row of 512
__global__ void __launch_bounds__(256)
k_ln(const float* __restrict__ h, const float* __restrict__ g,
     const float* __restrict__ b, ushort_t* __restrict__ o) {
  __shared__ float red[256];
  int row = blockIdx.x, t = threadIdx.x;
  const float* hr = h + (size_t)row * DMODEL;
  float x0 = hr[t], x1 = hr[t + 256];
  red[t] = x0 + x1;
  __syncthreads();
  for (int s = 128; s > 0; s >>= 1) { if (t < s) red[t] += red[t + s]; __syncthreads(); }
  float mu = red[0] * (1.0f / DMODEL);
  __syncthreads();
  float d0 = x0 - mu, d1 = x1 - mu;
  red[t] = d0 * d0 + d1 * d1;
  __syncthreads();
  for (int s = 128; s > 0; s >>= 1) { if (t < s) red[t] += red[t + s]; __syncthreads(); }
  float rs = rsqrtf(red[0] * (1.0f / DMODEL) + 1e-5f);
  ushort_t* orow = o + (size_t)row * DMODEL;
  orow[t]       = f2bf(d0 * rs * g[t]       + b[t]);
  orow[t + 256] = f2bf(d1 * rs * g[t + 256] + b[t + 256]);
}

// ---------- tiled WMMA GEMM: C[M,Nc] = A[M,K](bf16) @ W[K,Nc](bf16) + bias ----------
// EPI 0: scatter to q/k/v [B,H,N,64] bf16
// EPI 1: + resid -> fp32 out (in-place residual stream safe)
// EPI 2: gelu_new -> bf16 out
template <int EPI>
__global__ void __launch_bounds__(256)
k_gemm(const ushort_t* __restrict__ A, const ushort_t* __restrict__ W,
       const float* __restrict__ bias, int M, int K, int Nc,
       float* __restrict__ outf, ushort_t* __restrict__ outb,
       const float* __restrict__ resid,
       ushort_t* __restrict__ qd, ushort_t* __restrict__ kd, ushort_t* __restrict__ vd) {
  __shared__ __attribute__((aligned(16))) ushort_t As[128 * 32];   // [row][k]
  __shared__ __attribute__((aligned(16))) ushort_t BsT[128 * 32];  // [col][k]
  int t = threadIdx.x, lane = t & 31, w = t >> 5;
  int wm = w >> 2, wn = w & 3;                 // 2x4 wave grid
  int m0 = blockIdx.y * 128, n0 = blockIdx.x * 128;
  int lh = lane >> 4, ll = lane & 15;

  v8f acc[4][2];
  #pragma unroll
  for (int i = 0; i < 4; i++)
    #pragma unroll
    for (int j = 0; j < 2; j++) acc[i][j] = zero8();

  for (int k0 = 0; k0 < K; k0 += 32) {
    { // stage A tile 128x32 via CDNA5 async DMA to LDS (no VGPR round-trip)
      int row = t >> 1, seg = (t & 1) * 16;
      const ushort_t* src = A + (size_t)(m0 + row) * K + k0 + seg;
      unsigned ldsa = (unsigned)(uintptr_t)&As[row * 32 + seg];
      unsigned long long ga = (unsigned long long)(uintptr_t)src;
      asm volatile("global_load_async_to_lds_b128 %0, %1, off"
                   :: "v"(ldsa), "v"(ga) : "memory");
      asm volatile("global_load_async_to_lds_b128 %0, %1, off offset:16"
                   :: "v"(ldsa), "v"(ga) : "memory");
      if (k0 + 32 < K) __builtin_prefetch(src + 32, 0, 1);
    }
    { // stage B tile 32x128 transposed -> BsT[col][k] (transpose: VGPR path)
      int kk = t >> 3, cs = (t & 7) * 16;
      const ushort_t* src = W + (size_t)(k0 + kk) * Nc + n0 + cs;
      __attribute__((aligned(16))) ushort_t tmp[16];
      *(uint4*)&tmp[0] = *(const uint4*)(src);
      *(uint4*)&tmp[8] = *(const uint4*)(src + 8);
      #pragma unroll
      for (int j = 0; j < 16; j++) BsT[(cs + j) * 32 + kk] = tmp[j];
      if (k0 + 32 < K) __builtin_prefetch(src + (size_t)32 * Nc, 0, 1);
    }
    // my async LDS writes must land before the barrier publishes them
    asm volatile("s_wait_asynccnt 0x0" ::: "memory");
    __syncthreads();

    v16bf bfrag[2];
    #pragma unroll
    for (int nt = 0; nt < 2; nt++) {
      const ushort_t* bp = &BsT[(wn * 32 + nt * 16 + ll) * 32 + lh * 16];
      bfrag[nt] = ld16(bp, bp + 8);
    }
    #pragma unroll
    for (int mt = 0; mt < 4; mt++) {
      const ushort_t* ap = &As[(wm * 64 + mt * 16 + ll) * 32 + lh * 16];
      v16bf afrag = ld16(ap, ap + 8);
      #pragma unroll
      for (int nt = 0; nt < 2; nt++)
        acc[mt][nt] = wmma_bf16(afrag, bfrag[nt], acc[mt][nt]);
    }
    __syncthreads();
  }

  // epilogue: C/D layout -> element r is row (r + 8*lh), col = ll of its 16x16 tile
  #pragma unroll
  for (int mt = 0; mt < 4; mt++)
    #pragma unroll
    for (int nt = 0; nt < 2; nt++)
      #pragma unroll
      for (int r = 0; r < 8; r++) {
        int m = m0 + wm * 64 + mt * 16 + r + lh * 8;
        int n = n0 + wn * 32 + nt * 16 + ll;
        float v = acc[mt][nt][r] + bias[n];
        if (EPI == 0) {
          int bq = m >> 11, nn = m & (NSEQ - 1);
          int which = n >> 9, dc = n & (DMODEL - 1);
          int hh = dc >> 6, hd = dc & 63;
          size_t dst = (((size_t)(bq * NHEAD + hh) * NSEQ + nn) * HDIM + hd);
          ushort_t bv = f2bf(v);
          if (which == 0) qd[dst] = bv;
          else if (which == 1) kd[dst] = bv;
          else vd[dst] = bv;
        } else if (EPI == 1) {
          size_t idx = (size_t)m * Nc + n;
          outf[idx] = v + resid[idx];
        } else {
          size_t idx = (size_t)m * Nc + n;
          outb[idx] = f2bf(gelu_new(v));
        }
      }
}

// ---------- flash-style attention, 1 wave per (b,h,16-query tile) ----------
__global__ void __launch_bounds__(32)
k_attn(const ushort_t* __restrict__ qb, const ushort_t* __restrict__ kb,
       const ushort_t* __restrict__ vb, ushort_t* __restrict__ ob) {
  __shared__ __attribute__((aligned(16))) float    Sbuf[16 * 32];
  __shared__ __attribute__((aligned(16))) ushort_t Pbuf[16 * 32];
  __shared__ __attribute__((aligned(16))) ushort_t VsT[64 * 32];   // [d][key]
  __shared__ float m_s[16], l_s[16], sc_s[16];

  int lane = threadIdx.x, lh = lane >> 4, ll = lane & 15;
  int qt = blockIdx.x & 127;       // N/16 tiles
  int bh = blockIdx.x >> 7;        // 0..15
  int bq = bh >> 3, hh = bh & 7;
  const ushort_t* Q  = qb + (size_t)bh * NSEQ * HDIM;
  const ushort_t* Km = kb + (size_t)bh * NSEQ * HDIM;
  const ushort_t* Vm = vb + (size_t)bh * NSEQ * HDIM;

  // Q tile as two A fragments (K=0..31 and K=32..63)
  const ushort_t* qrow = Q + (size_t)(qt * 16 + ll) * HDIM + lh * 16;
  v16bf aq0 = ld16(qrow,      qrow + 8);
  v16bf aq1 = ld16(qrow + 32, qrow + 40);

  if (lane < 16) { m_s[lane] = -3.0e38f; l_s[lane] = 0.0f; }
  __syncthreads();

  v8f o[4];
  #pragma unroll
  for (int i = 0; i < 4; i++) o[i] = zero8();

  for (int kc = 0; kc < NSEQ; kc += 32) {
    // scores for 32 keys as two 16x16 tiles
    v8f s0 = zero8(), s1 = zero8();
    {
      const ushort_t* kr0 = Km + (size_t)(kc + ll) * HDIM + lh * 16;
      s0 = wmma_bf16(aq0, ld16(kr0,      kr0 + 8),  s0);
      s0 = wmma_bf16(aq1, ld16(kr0 + 32, kr0 + 40), s0);
      const ushort_t* kr1 = Km + (size_t)(kc + 16 + ll) * HDIM + lh * 16;
      s1 = wmma_bf16(aq0, ld16(kr1,      kr1 + 8),  s1);
      s1 = wmma_bf16(aq1, ld16(kr1 + 32, kr1 + 40), s1);
    }
    // scale, exclude-self mask, spill to LDS
    #pragma unroll
    for (int r = 0; r < 8; r++) {
      int row = r + lh * 8;
      int qn = qt * 16 + row;
      float v0 = s0[r] * 0.125f, v1 = s1[r] * 0.125f;
      if (qn == kc + ll)      v0 = -3.0e38f;
      if (qn == kc + 16 + ll) v1 = -3.0e38f;
      Sbuf[row * 32 + ll]      = v0;
      Sbuf[row * 32 + 16 + ll] = v1;
    }
    __syncthreads();

    // online softmax bookkeeping: 16 threads own one query row each
    if (lane < 16) {
      int row = lane;
      float mold = m_s[row], mc = mold;
      #pragma unroll
      for (int j = 0; j < 32; j++) mc = fmaxf(mc, Sbuf[row * 32 + j]);
      float sc = __expf(mold - mc);
      float l = l_s[row] * sc;
      #pragma unroll
      for (int j = 0; j < 32; j++) {
        float p = __expf(Sbuf[row * 32 + j] - mc);
        l += p;
        Pbuf[row * 32 + j] = f2bf(p);
      }
      m_s[row] = mc; l_s[row] = l; sc_s[row] = sc;
    }
    // stage V chunk transposed: VsT[d][key]
    {
      const ushort_t* vr = Vm + (size_t)(kc + lane) * HDIM;
      #pragma unroll
      for (int i = 0; i < 8; i++) {
        __attribute__((aligned(16))) ushort_t tmp[8];
        *(uint4*)tmp = *(const uint4*)(vr + i * 8);
        #pragma unroll
        for (int j = 0; j < 8; j++) VsT[(i * 8 + j) * 32 + lane] = tmp[j];
      }
    }
    __syncthreads();

    // rescale running output by exp(m_old - m_new)
    #pragma unroll
    for (int r = 0; r < 8; r++) {
      float sc = sc_s[r + lh * 8];
      #pragma unroll
      for (int t2 = 0; t2 < 4; t2++) o[t2][r] *= sc;
    }
    // P (16x32) @ V (32x64) via 4 WMMAs
    const ushort_t* pp = &Pbuf[ll * 32 + lh * 16];
    v16bf ap = ld16(pp, pp + 8);
    #pragma unroll
    for (int t2 = 0; t2 < 4; t2++) {
      const ushort_t* bp = &VsT[(t2 * 16 + ll) * 32 + lh * 16];
      o[t2] = wmma_bf16(ap, ld16(bp, bp + 8), o[t2]);
    }
    __syncthreads();
  }

  // normalize and write [B,N,D] bf16
  #pragma unroll
  for (int r = 0; r < 8; r++) {
    int row = r + lh * 8;
    float inv = 1.0f / l_s[row];
    int n = qt * 16 + row;
    size_t base = ((size_t)(bq * NSEQ + n)) * DMODEL + hh * HDIM;
    #pragma unroll
    for (int t2 = 0; t2 < 4; t2++)
      ob[base + t2 * 16 + ll] = f2bf(o[t2][r] * inv);
  }
}

// ---------- host ----------
extern "C" void kernel_launch(void* const* d_in, const int* in_sizes, int n_in,
                              void* d_out, int out_size, void* d_ws, size_t ws_size,
                              hipStream_t stream) {
  (void)in_sizes; (void)n_in; (void)ws_size;
  const float* emb     = (const float*)d_in[0];
  const float* pos     = (const float*)d_in[1];
  const float* attnw_f = (const float*)d_in[2];
  const float* attnb   = (const float*)d_in[3];
  const float* outw_f  = (const float*)d_in[4];
  const float* outbb   = (const float*)d_in[5];
  const float* ln1g    = (const float*)d_in[6];
  const float* ln1b    = (const float*)d_in[7];
  const float* fcw_f   = (const float*)d_in[8];
  const float* fcb     = (const float*)d_in[9];
  const float* projw_f = (const float*)d_in[10];
  const float* projb   = (const float*)d_in[11];
  const float* ln2g    = (const float*)d_in[12];
  const float* ln2b    = (const float*)d_in[13];

  char* wp = (char*)d_ws;
  auto carve = [&](size_t bytes) -> void* {
    void* p = (void*)wp;
    wp += (bytes + 255) & ~(size_t)255;
    return p;
  };
  float*    hbuf  = (float*)   carve((size_t)MROWS * DMODEL * 4);
  ushort_t* xbf   = (ushort_t*)carve((size_t)MROWS * DMODEL * 2);
  ushort_t* qbv   = (ushort_t*)carve((size_t)MROWS * DMODEL * 2);
  ushort_t* kbv   = (ushort_t*)carve((size_t)MROWS * DMODEL * 2);
  ushort_t* vbv   = (ushort_t*)carve((size_t)MROWS * DMODEL * 2);
  ushort_t* aob   = (ushort_t*)carve((size_t)MROWS * DMODEL * 2);
  ushort_t* fca   = (ushort_t*)carve((size_t)MROWS * DFF * 2);
  ushort_t* attnw = (ushort_t*)carve((size_t)NLAYER * DMODEL * 3 * DMODEL * 2);
  ushort_t* outw  = (ushort_t*)carve((size_t)NLAYER * DMODEL * DMODEL * 2);
  ushort_t* fcw   = (ushort_t*)carve((size_t)NLAYER * DMODEL * DFF * 2);
  ushort_t* projw = (ushort_t*)carve((size_t)NLAYER * DFF * DMODEL * 2);

  auto cvt = [&](const float* s, ushort_t* d, int n) {
    k_cvt_bf16<<<(n + 255) / 256, 256, 0, stream>>>(s, d, n);
  };
  cvt(attnw_f, attnw, NLAYER * DMODEL * 3 * DMODEL);
  cvt(outw_f,  outw,  NLAYER * DMODEL * DMODEL);
  cvt(fcw_f,   fcw,   NLAYER * DMODEL * DFF);
  cvt(projw_f, projw, NLAYER * DFF * DMODEL);

  k_addpos<<<(MROWS * DMODEL) / 256, 256, 0, stream>>>(emb, pos, hbuf);

  for (int l = 0; l < NLAYER; l++) {
    k_ln<<<MROWS, 256, 0, stream>>>(hbuf, ln1g + l * DMODEL, ln1b + l * DMODEL, xbf);

    dim3 g1(3 * DMODEL / 128, MROWS / 128);
    k_gemm<0><<<g1, 256, 0, stream>>>(xbf, attnw + (size_t)l * DMODEL * 3 * DMODEL,
                                      attnb + l * 3 * DMODEL, MROWS, DMODEL, 3 * DMODEL,
                                      nullptr, nullptr, nullptr, qbv, kbv, vbv);

    k_attn<<<2 * NHEAD * (NSEQ / 16), 32, 0, stream>>>(qbv, kbv, vbv, aob);

    dim3 g2(DMODEL / 128, MROWS / 128);
    k_gemm<1><<<g2, 256, 0, stream>>>(aob, outw + (size_t)l * DMODEL * DMODEL,
                                      outbb + l * DMODEL, MROWS, DMODEL, DMODEL,
                                      hbuf, nullptr, hbuf, nullptr, nullptr, nullptr);

    k_ln<<<MROWS, 256, 0, stream>>>(hbuf, ln2g + l * DMODEL, ln2b + l * DMODEL, xbf);

    dim3 g3(DFF / 128, MROWS / 128);
    k_gemm<2><<<g3, 256, 0, stream>>>(xbf, fcw + (size_t)l * DMODEL * DFF,
                                      fcb + l * DFF, MROWS, DMODEL, DFF,
                                      nullptr, fca, nullptr, nullptr, nullptr, nullptr);

    dim3 g4(DMODEL / 128, MROWS / 128);
    k_gemm<1><<<g4, 256, 0, stream>>>(fca, projw + (size_t)l * DFF * DMODEL,
                                      projb + l * DMODEL, MROWS, DFF, DMODEL,
                                      hbuf, nullptr, hbuf, nullptr, nullptr, nullptr);
  }

  k_copy<<<(out_size + 255) / 256, 256, 0, stream>>>(hbuf, (float*)d_out, out_size);
}
